// DilatedAttention_55654186222201
// MI455X (gfx1250) — matrix operand order repaired and checked by reading, not compiled, not run on hardware
//
#include <hip/hip_runtime.h>

typedef unsigned short u16;
typedef unsigned int   u32;
typedef __attribute__((ext_vector_type(4)))  unsigned int u32x4;
typedef __attribute__((ext_vector_type(4)))  int          v4i;
typedef __attribute__((ext_vector_type(4)))  float        f32x4;
typedef __attribute__((ext_vector_type(16))) __bf16       v16bf;
typedef __attribute__((ext_vector_type(8)))  float        v8f;

#define BATCH 2
#define SEQ   8192
#define HID   1024
#define NHEAD 16
#define DHEAD 64
#define WSEG  2048
#define RDIL  4
#define NPTS  512
#define NSEG  4
#define NOUT  3072   // fused [Q | K | V] columns

#if defined(__AMDGCN__) && __has_builtin(__builtin_amdgcn_global_load_async_to_lds_b128)
#define HAS_ASYNC 1
#else
#define HAS_ASYNC 0
#endif

union FragU { u32x4 q[2]; v16bf v; };

__device__ __forceinline__ u16 f2bf(float f) {
  u32 u = __float_as_uint(f);
  u32 r = u + 0x7FFFu + ((u >> 16) & 1u);
  return (u16)(r >> 16);
}

// Generic two-chunk (16B + 16B) fragment load.
__device__ __forceinline__ v16bf ld2(const u16* p1, const u16* p2) {
  FragU u;
  u.q[0] = *(const u32x4*)p1;
  u.q[1] = *(const u32x4*)p2;
  return u.v;
}

// 16B global -> LDS copy: async DMA on gfx1250, sync fallback elsewhere.
__device__ __forceinline__ void cp_async16(const u16* g, u16* l) {
#if HAS_ASYNC
  __builtin_amdgcn_global_load_async_to_lds_b128(
      (__attribute__((address_space(1))) v4i*)g,
      (__attribute__((address_space(3))) v4i*)l, 0, 0);
#else
  *(u32x4*)l = *(const u32x4*)g;
#endif
}

__device__ __forceinline__ void wait_async() {
#if HAS_ASYNC
#if __has_builtin(__builtin_amdgcn_s_wait_asynccnt)
  __builtin_amdgcn_s_wait_asynccnt(0);
#else
  asm volatile("s_wait_asynccnt 0" ::: "memory");
#endif
#endif
}

// ---------------------------------------------------------------------------
// Pack: X (f32) -> Xb (bf16, row-major [t][k]);
//       Wq/Wkv  -> WbT (bf16, TRANSPOSED [n=0..3071][k=0..1023]);
//       bq/bkv  -> biasc[3072] (f32)
// ---------------------------------------------------------------------------
__global__ void pack_kernel(const float* __restrict__ X,
                            const float* __restrict__ Wq,
                            const float* __restrict__ Wkv,
                            const float* __restrict__ bq,
                            const float* __restrict__ bkv,
                            u16* __restrict__ Xb,
                            u16* __restrict__ WbT,
                            float* __restrict__ biasc) {
  const int tid    = blockIdx.x * blockDim.x + threadIdx.x;
  const int stride = gridDim.x * blockDim.x;

  const long nvec = (long)BATCH * SEQ * HID / 4;
  for (long i = tid; i < nvec; i += stride) {
    f32x4 v = ((const f32x4*)X)[i];
    u32 lo = (u32)f2bf(v.x) | ((u32)f2bf(v.y) << 16);
    u32 hi = (u32)f2bf(v.z) | ((u32)f2bf(v.w) << 16);
    ((u32*)Xb)[2 * i]     = lo;
    ((u32*)Xb)[2 * i + 1] = hi;
  }

  const long nw = (long)NOUT * HID;
  for (long i = tid; i < nw; i += stride) {
    int n = (int)(i >> 10);      // output column 0..3071
    int k = (int)(i & 1023);     // reduction index
    float w = (n < HID) ? Wq[(long)k * HID + n]
                        : Wkv[(long)k * (2 * HID) + (n - HID)];
    WbT[i] = f2bf(w);
  }

  for (int i = tid; i < NOUT; i += stride)
    biasc[i] = (i < HID) ? bq[i] : bkv[i - HID];
}

// ---------------------------------------------------------------------------
// GEMM: Y[16384][3072] (bf16) = Xb[16384][1024] @ W + bias
// Block tile 128(M) x 128(N), 8 waves (2x4), each wave 64x32 = 4x2 WMMA tiles.
// A/B tiles double-buffered in LDS via async global->LDS DMA (ASYNCcnt).
// Row pitch 40 u16 (80B): keeps 16B alignment, breaks 64B bank stride.
// ---------------------------------------------------------------------------
__global__ __launch_bounds__(256) void gemm_bf16(const u16* __restrict__ Xb,
                                                 const u16* __restrict__ WbT,
                                                 const float* __restrict__ biasc,
                                                 u16* __restrict__ Y) {
  __shared__ u16 Ab[2][128 * 40];
  __shared__ u16 Bb[2][128 * 40];

  const int tid  = threadIdx.x;
  const int lane = tid & 31;
  const int wave = tid >> 5;
  const int ln   = lane & 15;
  const int lh   = lane >> 4;
  const int rowblk = blockIdx.y * 128;
  const int colblk = blockIdx.x * 128;
  const int wrow = (wave & 1) * 64;
  const int wcol = (wave >> 1) * 32;

  // Stage one 128x32 A tile + 128x32 B tile: 512 16B chunks each, 2+2/thread.
  auto stage = [&](int kk, int buf) {
#pragma unroll
    for (int i = 0; i < 2; ++i) {
      const int c = tid * 2 + i;       // 0..511
      const int row = c >> 2;          // 0..127
      const int part = c & 3;          // 8 u16 each
      cp_async16(Xb  + (long)(rowblk + row) * HID + kk + part * 8,
                 &Ab[buf][row * 40 + part * 8]);
      cp_async16(WbT + (long)(colblk + row) * HID + kk + part * 8,
                 &Bb[buf][row * 40 + part * 8]);
    }
  };

  v8f acc[4][2];
#pragma unroll
  for (int mt = 0; mt < 4; ++mt)
#pragma unroll
    for (int nt = 0; nt < 2; ++nt)
#pragma unroll
      for (int j = 0; j < 8; ++j) acc[mt][nt][j] = 0.0f;

  stage(0, 0);
  wait_async();
  __syncthreads();

  for (int kk = 0; kk < HID; kk += 32) {
    const int cur = (kk >> 5) & 1;
    if (kk + 32 < HID) stage(kk + 32, cur ^ 1);   // overlap DMA with WMMA

    v16bf a[4], b[2];
#pragma unroll
    for (int mt = 0; mt < 4; ++mt) {
      const u16* p = &Ab[cur][(wrow + mt * 16 + ln) * 40 + lh * 8];
      a[mt] = ld2(p, p + 16);                     // A layout
    }
#pragma unroll
    for (int nt = 0; nt < 2; ++nt) {
      const u16* p = &Bb[cur][(wcol + nt * 16 + ln) * 40 + lh * 8];
      b[nt] = ld2(p, p + 16);                     // B layout (table-order K)
    }
#pragma unroll
    for (int mt = 0; mt < 4; ++mt)
#pragma unroll
      for (int nt = 0; nt < 2; ++nt)
        acc[mt][nt] = __builtin_amdgcn_wmma_f32_16x16x32_bf16(
            false, a[mt], false, b[nt], (short)0, acc[mt][nt], false, false);

    wait_async();
    __syncthreads();
  }

#pragma unroll
  for (int nt = 0; nt < 2; ++nt) {
    const int col = colblk + wcol + nt * 16 + ln;
    const float bv = biasc[col];
#pragma unroll
    for (int mt = 0; mt < 4; ++mt) {
      const int r0 = rowblk + wrow + mt * 16 + lh * 8;
#pragma unroll
      for (int j = 0; j < 8; ++j)
        Y[(long)(r0 + j) * NOUT + col] = f2bf(acc[mt][nt][j] + bv);
    }
  }
}

// ---------------------------------------------------------------------------
// Dilated flash attention. Grid = B*NSEG*NHEAD*4 qchunks = 512 blocks.
// Each block: 128 query rows (8 waves x 16), loops all 512 keys of its
// (b, seg, head) in 64-key LDS stages, 32-key WMMA chunks, online softmax.
// K staged via async global->LDS DMA; V transposed through registers.
// ---------------------------------------------------------------------------
__global__ __launch_bounds__(256) void attn_kernel(const u16* __restrict__ Y,
                                                   float* __restrict__ out) {
  __shared__ u16 Kt[64 * 80];        // [key][d], pitch 80 (160B, 16B aligned)
  __shared__ u16 Vt[64 * 80];        // [d][key] transposed
  __shared__ u16 Ps[8 * 512];        // per-wave P staging: [16 rows][32 keys]

  const int tid  = threadIdx.x;
  const int lane = tid & 31;
  const int wave = tid >> 5;
  const int ln   = lane & 15;
  const int lh   = lane >> 4;

  const int bid = blockIdx.x;
  const int qc  = bid & 3;
  const int h   = (bid >> 2) & 15;
  const int seg = (bid >> 6) & 3;
  const int b   = (bid >> 8) & 1;
  const int shift = h & (RDIL - 1);
  const long seqbase = (long)b * SEQ + seg * WSEG;   // Y row base
  const int qbase = qc * 128 + wave * 16;
  const float scale = 0.125f;                         // 1/sqrt(64)

  // Q fragments (A operand), rows p = qbase + ln
  v16bf qa[2];
  {
    const int p = qbase + ln;
    const long grow = seqbase + 4 * p + shift;
    const u16* qp = Y + grow * NOUT + h * DHEAD;
    qa[0] = ld2(qp + lh * 8, qp + lh * 8 + 16);
    qa[1] = ld2(qp + 32 + lh * 8, qp + 32 + lh * 8 + 16);
  }

  float rm[8], rs[8];
  v8f o[4];
#pragma unroll
  for (int j = 0; j < 8; ++j) { rm[j] = -1e30f; rs[j] = 0.0f; }
#pragma unroll
  for (int nt = 0; nt < 4; ++nt)
#pragma unroll
    for (int j = 0; j < 8; ++j) o[nt][j] = 0.0f;

  for (int sc = 0; sc < 8; ++sc) {                 // 8 x 64-key stages
    __syncthreads();
    {                                              // cooperative K/V staging
      const int kr = tid >> 2;                     // 0..63 key row
      const int part = tid & 3;                    // 16 d-elements each
      const int p = sc * 64 + kr;
      const long grow = seqbase + 4 * p + shift;
      const u16* krow = Y + grow * NOUT + HID + h * DHEAD + part * 16;
      cp_async16(krow,     &Kt[kr * 80 + part * 16]);
      cp_async16(krow + 8, &Kt[kr * 80 + part * 16 + 8]);

      const u16* vrow = Y + grow * NOUT + 2 * HID + h * DHEAD + part * 16;
      u16 vv[16];
      *(u32x4*)&vv[0] = *(const u32x4*)vrow;
      *(u32x4*)&vv[8] = *(const u32x4*)(vrow + 8);
#pragma unroll
      for (int e = 0; e < 16; ++e)
        Vt[(part * 16 + e) * 80 + kr] = vv[e];     // transpose into [d][key]
    }
    wait_async();
    __syncthreads();

    for (int hc = 0; hc < 2; ++hc) {               // two 32-key chunks
      // ---- scores: S = Q @ K^T, two 16-key tiles ----
      v8f s[2];
#pragma unroll
      for (int sub = 0; sub < 2; ++sub) {
        const int kr2 = hc * 32 + sub * 16 + ln;   // key row (= B column)
        const u16* kp = &Kt[kr2 * 80];
        v16bf kb0 = ld2(kp + lh * 8, kp + lh * 8 + 16);            // d 0..31
        v16bf kb1 = ld2(kp + 32 + lh * 8, kp + 32 + lh * 8 + 16);  // d 32..63
        v8f z;
#pragma unroll
        for (int j = 0; j < 8; ++j) z[j] = 0.0f;
        z = __builtin_amdgcn_wmma_f32_16x16x32_bf16(false, qa[0], false, kb0,
                                                    (short)0, z, false, false);
        z = __builtin_amdgcn_wmma_f32_16x16x32_bf16(false, qa[1], false, kb1,
                                                    (short)0, z, false, false);
        s[sub] = z;
      }

      // ---- online softmax over this 32-key chunk ----
      float p0[8], p1[8];
#pragma unroll
      for (int j = 0; j < 8; ++j) {
        float v0 = s[0][j] * scale;
        float v1 = s[1][j] * scale;
        float tm = fmaxf(v0, v1);
#pragma unroll
        for (int m = 1; m < 16; m <<= 1) tm = fmaxf(tm, __shfl_xor(tm, m, 32));
        float nm = fmaxf(rm[j], tm);
        float corr = __expf(rm[j] - nm);
        float e0 = __expf(v0 - nm);
        float e1 = __expf(v1 - nm);
        float ts = e0 + e1;
#pragma unroll
        for (int m = 1; m < 16; m <<= 1) ts += __shfl_xor(ts, m, 32);
        rs[j] = rs[j] * corr + ts;
        rm[j] = nm;
#pragma unroll
        for (int nt = 0; nt < 4; ++nt) o[nt][j] *= corr;
        p0[j] = e0; p1[j] = e1;
      }

      // ---- P (C layout) -> LDS -> A layout fragment ----
      u16* pw = &Ps[wave * 512];
#pragma unroll
      for (int j = 0; j < 8; ++j) {
        const int m = lh * 8 + j;
        pw[m * 32 + ln]      = f2bf(p0[j]);
        pw[m * 32 + 16 + ln] = f2bf(p1[j]);
      }
      const u16* pp = &Ps[wave * 512 + ln * 32];
      v16bf pa = ld2(pp + lh * 8, pp + lh * 8 + 16);

      // ---- O += P @ V ----
#pragma unroll
      for (int nt = 0; nt < 4; ++nt) {
        const u16* vp = &Vt[(nt * 16 + ln) * 80 + hc * 32];
        v16bf vb = ld2(vp + lh * 8, vp + lh * 8 + 16);
        o[nt] = __builtin_amdgcn_wmma_f32_16x16x32_bf16(false, pa, false, vb,
                                                        (short)0, o[nt], false, false);
      }
    }
  }

  // ---- normalize and scatter to output ----
#pragma unroll
  for (int nt = 0; nt < 4; ++nt) {
#pragma unroll
    for (int j = 0; j < 8; ++j) {
      const int p = qbase + lh * 8 + j;
      const long t = (long)seg * WSEG + 4 * p + shift;
      out[((long)b * SEQ + t) * HID + h * DHEAD + nt * 16 + ln] = o[nt][j] / rs[j];
    }
  }
}

// ---------------------------------------------------------------------------
extern "C" void kernel_launch(void* const* d_in, const int* in_sizes, int n_in,
                              void* d_out, int out_size, void* d_ws, size_t ws_size,
                              hipStream_t stream) {
  const float* X   = (const float*)d_in[0];
  const float* Wq  = (const float*)d_in[1];
  const float* bq  = (const float*)d_in[2];
  const float* Wkv = (const float*)d_in[3];
  const float* bkv = (const float*)d_in[4];
  float* out = (float*)d_out;

  char* ws = (char*)d_ws;
  u16*   Xb    = (u16*)(ws);                          // 33,554,432 B
  u16*   WbT   = (u16*)(ws + 33554432);               //  6,291,456 B
  float* biasc = (float*)(ws + 39845888);             //     12,288 B
  u16*   Y     = (u16*)(ws + 39858176);               // 100,663,296 B

  (void)hipMemsetAsync(d_out, 0, (size_t)out_size * sizeof(float), stream);

  pack_kernel<<<2048, 256, 0, stream>>>(X, Wq, Wkv, bq, bkv, Xb, WbT, biasc);

  dim3 gg(NOUT / 128, (BATCH * SEQ) / 128);           // 24 x 128
  gemm_bf16<<<gg, 256, 0, stream>>>(Xb, WbT, biasc, Y);

  attn_kernel<<<BATCH * NSEG * NHEAD * 4, 256, 0, stream>>>(Y, out);
}
